// SS2DBottleneck_77120432767199
// MI455X (gfx1250) — matrix operand order repaired and checked by reading, not compiled, not run on hardware
//
#include <hip/hip_runtime.h>
#include <math.h>
#include <stdint.h>

// ---------------------------------------------------------------------------
// CDNA5 / gfx1250 SS2D bottleneck. All dense linear algebra goes through
// v_wmma_f32_16x16x32_f16 (wave32). Sequential selective scan is VALU with
// async double-buffered LDS broadcast (ASYNCcnt path when available).
// ---------------------------------------------------------------------------

typedef __attribute__((ext_vector_type(16))) _Float16 v16h;
typedef __attribute__((ext_vector_type(8)))  float    v8f;

#define DEV __device__ __forceinline__

// Static problem sizes
#define BATCH   16
#define C_IN    256
#define C1      512
#define H1      56
#define HW1     3136      // 56*56
#define HP      28
#define LQ      784       // 28*28
#define DI      1024
#define NS      16
#define RK      32
#define KD      4
#define M_BL    (BATCH * LQ)   // 12544

// ---------------------------------------------------------------------------
// Async global->LDS copy (gfx1250 GLOBAL_LOAD_ASYNC_TO_LDS_B32, ASYNCcnt).
// Builtin signature (per hipcc diagnostic): (int as1*, int as3*, imm, imm).
// Address-space conversion goes through integers: generic shared-memory
// addresses truncate to the 32-bit LDS offset (ISA 10.2 aperture rule).
// ---------------------------------------------------------------------------
#if __has_builtin(__builtin_amdgcn_global_load_async_to_lds_b32)
#define HAVE_ASYNC_LDS 1
typedef __attribute__((address_space(1))) int g_as1_i32;
typedef __attribute__((address_space(3))) int l_as3_i32;
DEV void async_copy_f32(float* lds_dst, const float* gsrc) {
  __builtin_amdgcn_global_load_async_to_lds_b32(
      (g_as1_i32*)(uintptr_t)gsrc,
      (l_as3_i32*)(uint32_t)(uintptr_t)lds_dst,
      0, 0);
}
#else
DEV void async_copy_f32(float* lds_dst, const float* gsrc) { *lds_dst = *gsrc; }
#endif

DEV void wait_async0() {
#if defined(HAVE_ASYNC_LDS)
#if __has_builtin(__builtin_amdgcn_s_wait_asynccnt)
  __builtin_amdgcn_s_wait_asynccnt(0);
#else
  asm volatile("s_wait_asynccnt 0" ::: "memory");
#endif
#endif
}

// Direction index map: sequence position l -> memory position (h*28+w)
DEV int smap(int k, int l) {
  int m = (k & 2) ? (LQ - 1 - l) : l;
  if (k & 1) { int h = m % HP, w = m / HP; m = h * HP + w; }
  return m;
}

// ---------------------------------------------------------------------------
// Generic WMMA 16x16x32 f16 GEMM tile engine.
// Block = 128 threads (4 waves). Wave w computes rows [bx*64+w*16, +16),
// cols [by*64, +64) via 4 accumulators. Loaders are lambdas (row,k)->f16,
// (k,col)->f16; store (row,col,f32acc).
// ---------------------------------------------------------------------------
template <typename AL, typename BL, typename ST>
DEV void wmma_tile(const AL& al, const BL& bl, const ST& st, int K) {
  const int lane = threadIdx.x & 31;
  const int wave = threadIdx.x >> 5;
  const int m0   = blockIdx.x * 64 + wave * 16;
  const int n0   = blockIdx.y * 64;
  const int arow = m0 + (lane & 15);
  const int kA   = (lane >> 4) << 3;   // +0 / +8
  const int kB   = (lane >> 4) << 4;   // +0 / +16
  const int nc   = lane & 15;

  v8f c0 = {}, c1 = {}, c2 = {}, c3 = {};
  for (int kk = 0; kk < K; kk += 32) {
    v16h a;
#pragma unroll
    for (int i = 0; i < 16; ++i) {
      const int v  = i >> 1;
      const int kd = kA + ((v >> 2) << 4) + ((v & 3) << 1) + (i & 1);
      a[i] = al(arow, kk + kd);
    }
    v16h b0, b1, b2, b3;
#pragma unroll
    for (int i = 0; i < 16; ++i) {
      const int kd = kk + kA + ((i >> 3) << 4) + (i & 7);
      b0[i] = bl(kd, n0 + nc);
      b1[i] = bl(kd, n0 + 16 + nc);
      b2[i] = bl(kd, n0 + 32 + nc);
      b3[i] = bl(kd, n0 + 48 + nc);
    }
    c0 = __builtin_amdgcn_wmma_f32_16x16x32_f16(false, a, false, b0, (short)0, c0, false, false);
    c1 = __builtin_amdgcn_wmma_f32_16x16x32_f16(false, a, false, b1, (short)0, c1, false, false);
    c2 = __builtin_amdgcn_wmma_f32_16x16x32_f16(false, a, false, b2, (short)0, c2, false, false);
    c3 = __builtin_amdgcn_wmma_f32_16x16x32_f16(false, a, false, b3, (short)0, c3, false, false);
  }
  const int rb = m0 + ((lane >> 4) << 3);
#pragma unroll
  for (int j = 0; j < 8; ++j) {
    st(rb + j, n0 +      nc, c0[j]);
    st(rb + j, n0 + 16 + nc, c1[j]);
    st(rb + j, n0 + 32 + nc, c2[j]);
    st(rb + j, n0 + 48 + nc, c3[j]);
  }
}

// ---------------------------------------------------------------------------
// Prep kernels
// ---------------------------------------------------------------------------
__global__ void k_cvt(const float* __restrict__ s, _Float16* __restrict__ d, int n) {
  int i = blockIdx.x * 256 + threadIdx.x;
  if (i < n) d[i] = (_Float16)s[i];
}

__global__ void k_bnprep(const float* __restrict__ cb, const float* __restrict__ g,
                         const float* __restrict__ b,  const float* __restrict__ m,
                         const float* __restrict__ v,  float* __restrict__ alpha,
                         float* __restrict__ beta) {
  int c = blockIdx.x * 256 + threadIdx.x;
  if (c < C1) {
    float s = g[c] * rsqrtf(v[c] + 1e-5f);
    alpha[c] = s;
    beta[c]  = (cb[c] - m[c]) * s + b[c];
  }
}

__global__ void k_zero(float* __restrict__ p, int n) {
  int i = blockIdx.x * 256 + threadIdx.x;
  if (i < n) p[i] = 0.f;
}

// ---------------------------------------------------------------------------
// Stage 1: conv1 (1x1) as GEMM M=B*3136, N=512, K=256; epilogue BN+ReLU, f16 NCHW
// ---------------------------------------------------------------------------
__global__ void k_conv1(const float* __restrict__ x, const _Float16* __restrict__ w1,
                        const float* __restrict__ al_, const float* __restrict__ be_,
                        _Float16* __restrict__ h1) {
  auto A = [&](int mm, int k) -> _Float16 {
    int n = mm / HW1, pos = mm % HW1;
    return (_Float16)x[((size_t)(n * C_IN) + k) * HW1 + pos];
  };
  auto B = [&](int k, int o) -> _Float16 { return w1[(size_t)o * C_IN + k]; };
  auto S = [&](int mm, int o, float v) {
    int n = mm / HW1, pos = mm % HW1;
    float r = fmaxf(al_[o] * v + be_[o], 0.f);
    h1[((size_t)(n * C1) + o) * HW1 + pos] = (_Float16)r;
  };
  wmma_tile(A, B, S, C_IN);
}

// Stage 2: 2x2 maxpool -> NHWC f32 (kept for residual)
__global__ void k_pool(const _Float16* __restrict__ h1, float* __restrict__ xb) {
  int idx = blockIdx.x * 256 + threadIdx.x;
  if (idx >= M_BL * C1) return;
  int c = idx & (C1 - 1);
  int t = idx >> 9;
  int l = t % LQ, b = t / LQ;
  int h = l / HP, w = l % HP;
  const _Float16* src = h1 + ((size_t)(b * C1 + c)) * HW1 + (h * 2) * H1 + w * 2;
  float m0 = (float)src[0], m1 = (float)src[1];
  float m2 = (float)src[H1], m3 = (float)src[H1 + 1];
  xb[idx] = fmaxf(fmaxf(m0, m1), fmaxf(m2, m3));
}

// Stage 3: LayerNorm over C=512 rows -> f16
__global__ void k_ln512(const float* __restrict__ xin, const float* __restrict__ g,
                        const float* __restrict__ bta, _Float16* __restrict__ out) {
  const int row = blockIdx.x;
  const int tid = threadIdx.x;
  const float* p = xin + (size_t)row * C1;
  float v0 = p[tid], v1 = p[tid + 256];
  __shared__ float s1[256], s2[256];
  s1[tid] = v0 + v1;
  s2[tid] = v0 * v0 + v1 * v1;
  __syncthreads();
  for (int o = 128; o > 0; o >>= 1) {
    if (tid < o) { s1[tid] += s1[tid + o]; s2[tid] += s2[tid + o]; }
    __syncthreads();
  }
  float mu = s1[0] * (1.f / C1);
  float var = s2[0] * (1.f / C1) - mu * mu;
  float rs = rsqrtf(var + 1e-5f);
  out[(size_t)row * C1 + tid]       = (_Float16)((v0 - mu) * rs * g[tid] + bta[tid]);
  out[(size_t)row * C1 + tid + 256] = (_Float16)((v1 - mu) * rs * g[tid + 256] + bta[tid + 256]);
}

// Stage 4: in_proj GEMM M=12544, N=2048, K=512; split epilogue x->NCHW, z->NHWC
__global__ void k_inproj(const _Float16* __restrict__ xn, const _Float16* __restrict__ wip,
                         _Float16* __restrict__ xx, _Float16* __restrict__ z) {
  auto A = [&](int mm, int k) -> _Float16 { return xn[(size_t)mm * C1 + k]; };
  auto B = [&](int k, int e) -> _Float16 { return wip[(size_t)e * C1 + k]; };
  auto S = [&](int mm, int e, float v) {
    int b = mm / LQ, l = mm % LQ;
    if (e < DI) xx[((size_t)(b * DI) + e) * LQ + l] = (_Float16)v;
    else        z[(size_t)mm * DI + (e - DI)] = (_Float16)v;
  };
  wmma_tile(A, B, S, C1);
}

// Stage 5: depthwise 3x3 + bias + SiLU
__global__ void k_dwconv(const _Float16* __restrict__ xx, const float* __restrict__ w,
                         const float* __restrict__ bias, _Float16* __restrict__ xc) {
  int idx = blockIdx.x * 256 + threadIdx.x;
  if (idx >= BATCH * DI * LQ) return;
  int l = idx % LQ;
  int bd = idx / LQ;
  int d = bd & (DI - 1);
  int h = l / HP, ww = l % HP;
  float acc = bias[d];
  const _Float16* base = xx + (size_t)bd * LQ;
#pragma unroll
  for (int dy = 0; dy < 3; ++dy) {
    int hh = h + dy - 1;
    if ((unsigned)hh >= (unsigned)HP) continue;
#pragma unroll
    for (int dx = 0; dx < 3; ++dx) {
      int wc = ww + dx - 1;
      if ((unsigned)wc >= (unsigned)HP) continue;
      acc = fmaf((float)base[hh * HP + wc], w[d * 9 + dy * 3 + dx], acc);
    }
  }
  float sg = acc / (1.f + expf(-acc));
  xc[idx] = (_Float16)sg;
}

// Stage 6: x_proj GEMM per (b,dir): (64 x 784) = W[k](64x1024) * xs (1024x784)
// with direction remap fused into B-loader.
__global__ void k_xproj(const _Float16* __restrict__ wxp, const _Float16* __restrict__ xc,
                        float* __restrict__ xdbl) {
  const int z = blockIdx.z;
  const int b = z >> 2, kd = z & 3;
  auto A = [&](int c, int d) -> _Float16 { return wxp[((size_t)(kd * 64) + c) * DI + d]; };
  auto B = [&](int d, int l) -> _Float16 {
    return (l < LQ) ? xc[((size_t)(b * DI) + d) * LQ + smap(kd, l)] : (_Float16)0.f;
  };
  auto S = [&](int c, int l, float v) {
    if (l < LQ && c < 64) xdbl[(((size_t)(b * KD + kd)) * 64 + c) * LQ + l] = v;
  };
  wmma_tile(A, B, S, DI);
}

// ---------------------------------------------------------------------------
// Stage 7: fused dt_proj + softplus + selective scan + D*u, scatter-add into
// direction-combined accumulator ysum (B, L, DI).
// Per-step 64-float broadcast (32 dts + 16 B + 16 C) is double-buffered in LDS
// via async global->LDS copies: step l+1 is in flight while step l computes.
// ---------------------------------------------------------------------------
__global__ void k_scan(const float* __restrict__ xdbl, const _Float16* __restrict__ xc,
                       const float* __restrict__ dtw, const float* __restrict__ dtb,
                       const float* __restrict__ alogs, const float* __restrict__ Ds,
                       float* __restrict__ ysum) {
  const int tid = threadIdx.x;
  const int d = blockIdx.x * 128 + tid;
  const int kd = blockIdx.y, b = blockIdx.z;
  const int row = kd * DI + d;

  float w[RK];
#pragma unroll
  for (int r = 0; r < RK; ++r) w[r] = dtw[(size_t)row * RK + r];
  const float bias = dtb[row];
  float a[NS], st[NS];
#pragma unroll
  for (int n = 0; n < NS; ++n) { a[n] = -expf(alogs[(size_t)row * NS + n]); st[n] = 0.f; }
  const float Dv = Ds[row];

  __shared__ float sh[2][64];  // double-buffered: 32 dts + 16 B + 16 C per step
  const float* xd = xdbl + ((size_t)(b * KD + kd)) * 64 * LQ;
  const _Float16* xcr = xc + ((size_t)(b * DI) + d) * LQ;

  // Prologue: stage step 0.
  if (tid < 64) async_copy_f32(&sh[0][tid], &xd[(size_t)tid * LQ]);
  wait_async0();
  __syncthreads();

  for (int l = 0; l < LQ; ++l) {
    const int cur = l & 1;
    // Kick off the async stage of step l+1 into the other buffer. Safe: the
    // end-of-previous-iteration barrier guarantees everyone is done with it.
    if (l + 1 < LQ && tid < 64)
      async_copy_f32(&sh[cur ^ 1][tid], &xd[(size_t)tid * LQ + l + 1]);

    const float* s = sh[cur];
    float dp = bias;
#pragma unroll
    for (int r = 0; r < RK; ++r) dp = fmaf(w[r], s[r], dp);
    float delta = (dp > 20.f) ? dp : log1pf(expf(dp));
    int mpos = smap(kd, l);
    __builtin_prefetch(&xcr[smap(kd, (l + 8 < LQ) ? (l + 8) : (LQ - 1))], 0, 3);
    float u = (float)xcr[mpos];
    float du = delta * u;
    float y = 0.f;
#pragma unroll
    for (int n = 0; n < NS; ++n) {
      st[n] = expf(delta * a[n]) * st[n] + du * s[32 + n];
      y = fmaf(st[n], s[48 + n], y);
    }
    atomicAdd(&ysum[((size_t)(b * LQ) + mpos) * DI + d], y + Dv * u);

    wait_async0();      // next step's LDS data has landed (issuing waves)
    __syncthreads();    // publish to all waves; also fences reads of `cur`
  }
}

// Stage 8: out-LayerNorm over DI, gate with SiLU(z) -> f16
__global__ void k_ln_gate(const float* __restrict__ y, const float* __restrict__ g,
                          const float* __restrict__ bta, const _Float16* __restrict__ z,
                          _Float16* __restrict__ out) {
  const int row = blockIdx.x;
  const int tid = threadIdx.x;
  const float* p = y + (size_t)row * DI;
  float v[4];
  float s = 0.f, q = 0.f;
#pragma unroll
  for (int j = 0; j < 4; ++j) { v[j] = p[tid + j * 256]; s += v[j]; q += v[j] * v[j]; }
  __shared__ float s1[256], s2[256];
  s1[tid] = s; s2[tid] = q;
  __syncthreads();
  for (int o = 128; o > 0; o >>= 1) {
    if (tid < o) { s1[tid] += s1[tid + o]; s2[tid] += s2[tid + o]; }
    __syncthreads();
  }
  float mu = s1[0] * (1.f / DI);
  float var = s2[0] * (1.f / DI) - mu * mu;
  float rs = rsqrtf(var + 1e-5f);
#pragma unroll
  for (int j = 0; j < 4; ++j) {
    int c = tid + j * 256;
    float ln = (v[j] - mu) * rs * g[c] + bta[c];
    float zz = (float)z[(size_t)row * DI + c];
    float sz = zz / (1.f + expf(-zz));
    out[(size_t)row * DI + c] = (_Float16)(ln * sz);
  }
}

// Stage 9: out_proj GEMM M=12544, N=512, K=1024; epilogue residual add -> f16 NHWC
__global__ void k_outproj(const _Float16* __restrict__ y2, const _Float16* __restrict__ wop,
                          const float* __restrict__ xb, _Float16* __restrict__ h2) {
  auto A = [&](int mm, int k) -> _Float16 { return y2[(size_t)mm * DI + k]; };
  auto B = [&](int k, int c) -> _Float16 { return wop[(size_t)c * DI + k]; };
  auto S = [&](int mm, int c, float v) {
    h2[(size_t)mm * C1 + c] = (_Float16)(xb[(size_t)mm * C1 + c] + v);
  };
  wmma_tile(A, B, S, DI);
}

// Stage 10: conv2 3x3 implicit GEMM M=12544, N=512, K=4608; BN+ReLU -> f32 NCHW
__global__ void k_conv2(const _Float16* __restrict__ h2, const _Float16* __restrict__ w2,
                        const float* __restrict__ al_, const float* __restrict__ be_,
                        float* __restrict__ out) {
  auto A = [&](int mm, int k) -> _Float16 {
    int b = mm / LQ, pos = mm % LQ;
    int h = pos / HP, w = pos % HP;
    int t = k >> 9;          // tap 0..8
    int c = k & 511;
    int hh = h + t / 3 - 1, ww = w + t % 3 - 1;
    if ((unsigned)hh < (unsigned)HP && (unsigned)ww < (unsigned)HP)
      return h2[((size_t)(b * LQ) + hh * HP + ww) * C1 + c];
    return (_Float16)0.f;
  };
  auto B = [&](int k, int o) -> _Float16 {
    int t = k >> 9, c = k & 511;
    return w2[(size_t)o * (C1 * 9) + c * 9 + t];
  };
  auto S = [&](int mm, int o, float v) {
    int b = mm / LQ, pos = mm % LQ;
    float r = fmaxf(al_[o] * v + be_[o], 0.f);
    out[((size_t)(b * C1) + o) * LQ + pos] = r;
  };
  wmma_tile(A, B, S, C1 * 9);
}

// ---------------------------------------------------------------------------
extern "C" void kernel_launch(void* const* d_in, const int* in_sizes, int n_in,
                              void* d_out, int out_size, void* d_ws, size_t ws_size,
                              hipStream_t stream) {
  (void)in_sizes; (void)n_in; (void)out_size; (void)ws_size;
  const float* x      = (const float*)d_in[0];
  const float* w1     = (const float*)d_in[1];
  const float* c1b    = (const float*)d_in[2];
  const float* bn1g   = (const float*)d_in[3];
  const float* bn1b   = (const float*)d_in[4];
  const float* bn1m   = (const float*)d_in[5];
  const float* bn1v   = (const float*)d_in[6];
  const float* lng    = (const float*)d_in[7];
  const float* lnb    = (const float*)d_in[8];
  const float* wip    = (const float*)d_in[9];
  const float* dww    = (const float*)d_in[10];
  const float* dwb    = (const float*)d_in[11];
  const float* wxp    = (const float*)d_in[12];
  const float* dtw    = (const float*)d_in[13];
  const float* dtb    = (const float*)d_in[14];
  const float* alogs  = (const float*)d_in[15];
  const float* Ds     = (const float*)d_in[16];
  const float* ong    = (const float*)d_in[17];
  const float* onb    = (const float*)d_in[18];
  const float* wop    = (const float*)d_in[19];
  const float* w2     = (const float*)d_in[20];
  const float* c2b    = (const float*)d_in[21];
  const float* bn2g   = (const float*)d_in[22];
  const float* bn2b   = (const float*)d_in[23];
  const float* bn2m   = (const float*)d_in[24];
  const float* bn2v   = (const float*)d_in[25];
  float* out = (float*)d_out;

  // Bump allocator over d_ws with aliasing of dead regions.
  char* p = (char*)d_ws;
  auto alloc = [&](size_t bytes) -> void* {
    void* r = (void*)p;
    p += (bytes + 255) & ~(size_t)255;
    return r;
  };
  _Float16* w1h  = (_Float16*)alloc((size_t)C1 * C_IN * 2);
  _Float16* wiph = (_Float16*)alloc((size_t)2 * DI * C1 * 2);
  _Float16* wxph = (_Float16*)alloc((size_t)KD * 64 * DI * 2);
  _Float16* woph = (_Float16*)alloc((size_t)C1 * DI * 2);
  _Float16* w2h  = (_Float16*)alloc((size_t)C1 * C1 * 9 * 2);
  float* alpha1  = (float*)alloc(C1 * 4);
  float* beta1   = (float*)alloc(C1 * 4);
  float* alpha2  = (float*)alloc(C1 * 4);
  float* beta2   = (float*)alloc(C1 * 4);
  // region A: h1 (f16, 51.4MB) later reused as ysum (f32, 51.4MB)
  void* regA = alloc((size_t)BATCH * C1 * HW1 * 2);
  _Float16* h1 = (_Float16*)regA;
  float* ysum  = (float*)regA;
  float* xb    = (float*)alloc((size_t)M_BL * C1 * 4);
  // region B: xn (f16) later reused as h2 (same size)
  void* regB = alloc((size_t)M_BL * C1 * 2);
  _Float16* xn = (_Float16*)regB;
  _Float16* h2 = (_Float16*)regB;
  // region C: xx (f16 NCHW) later reused as y2 (same size)
  void* regC = alloc((size_t)BATCH * DI * LQ * 2);
  _Float16* xx = (_Float16*)regC;
  _Float16* y2 = (_Float16*)regC;
  _Float16* zg = (_Float16*)alloc((size_t)M_BL * DI * 2);
  _Float16* xc = (_Float16*)alloc((size_t)BATCH * DI * LQ * 2);
  float* xdbl  = (float*)alloc((size_t)BATCH * KD * 64 * LQ * 4);

  // --- prep: weight f16 conversion + BN folding ---
  auto cvt = [&](const float* s, _Float16* d, int n) {
    k_cvt<<<dim3((n + 255) / 256), dim3(256), 0, stream>>>(s, d, n);
  };
  cvt(w1,  w1h,  C1 * C_IN);
  cvt(wip, wiph, 2 * DI * C1);
  cvt(wxp, wxph, KD * 64 * DI);
  cvt(wop, woph, C1 * DI);
  cvt(w2,  w2h,  C1 * C1 * 9);
  k_bnprep<<<dim3(2), dim3(256), 0, stream>>>(c1b, bn1g, bn1b, bn1m, bn1v, alpha1, beta1);
  k_bnprep<<<dim3(2), dim3(256), 0, stream>>>(c2b, bn2g, bn2b, bn2m, bn2v, alpha2, beta2);

  // --- pipeline ---
  k_conv1<<<dim3((BATCH * HW1) / 64, C1 / 64), dim3(128), 0, stream>>>(x, w1h, alpha1, beta1, h1);
  k_pool<<<dim3((M_BL * C1) / 256), dim3(256), 0, stream>>>(h1, xb);
  k_ln512<<<dim3(M_BL), dim3(256), 0, stream>>>(xb, lng, lnb, xn);
  k_inproj<<<dim3(M_BL / 64, (2 * DI) / 64), dim3(128), 0, stream>>>(xn, wiph, xx, zg);
  k_dwconv<<<dim3((BATCH * DI * LQ) / 256), dim3(256), 0, stream>>>(xx, dww, dwb, xc);
  k_xproj<<<dim3(1, (LQ + 63) / 64, BATCH * KD), dim3(128), 0, stream>>>(wxph, xc, xdbl);
  k_zero<<<dim3((M_BL * DI) / 256), dim3(256), 0, stream>>>(ysum, M_BL * DI);
  k_scan<<<dim3(DI / 128, KD, BATCH), dim3(128), 0, stream>>>(xdbl, xc, dtw, dtb, alogs, Ds, ysum);
  k_ln_gate<<<dim3(M_BL), dim3(256), 0, stream>>>(ysum, ong, onb, zg, y2);
  k_outproj<<<dim3(M_BL / 64, C1 / 64), dim3(128), 0, stream>>>(y2, woph, xb, h2);
  k_conv2<<<dim3(M_BL / 64, C1 / 64), dim3(128), 0, stream>>>(h2, w2h, alpha2, beta2, out);
}
